// AttentionLayer2_1657857376449
// MI455X (gfx1250) — matrix operand order repaired and checked by reading, not compiled, not run on hardware
//
#include <hip/hip_runtime.h>

typedef __attribute__((ext_vector_type(16))) __bf16  v16bf;
typedef __attribute__((ext_vector_type(8)))  float   v8f;

#define B_   8
#define S_   2048
#define D_   1024
#define QT   64          // queries per block
#define KT   128         // keys per panel
#define NKT  (S_ / KT)   // 16 panels
#define DC   256         // D streamed in four 256-wide chunks (64 KiB each)
#define NCH  (NKT * 4)   // 64 chunks per pass
#define THREADS 256
#define KBUF_USH (KT * DC)   // 32768 ushorts = 64 KiB per K buffer

#define XBF_BYTES ((size_t)B_ * S_ * D_ * 2)   // 32 MiB bf16 copy of x in ws

union FragU {
  unsigned short us[16];
  uint4 u4[2];
  v16bf v;
};

static __device__ __forceinline__ unsigned short f32_to_bf16(float f) {
  unsigned int u = __float_as_uint(f);
  u += 0x7fffu + ((u >> 16) & 1u);       // round-to-nearest-even
  return (unsigned short)(u >> 16);
}

// ---- CDNA5 async global->LDS copy (ASYNCcnt path) ----
static __device__ __forceinline__ void async_cp16(unsigned lds_off,
                                                  unsigned long long gaddr) {
  asm volatile("global_load_async_to_lds_b128 %0, %1, off"
               :: "v"(lds_off), "v"(gaddr) : "memory");
}
static __device__ __forceinline__ void wait_async0() {
  asm volatile("s_wait_asynccnt 0" ::: "memory");
}
static __device__ __forceinline__ void wait_async16() {
  asm volatile("s_wait_asynccnt 16" ::: "memory");   // older chunk done, newer in flight
}

static __device__ __forceinline__ v8f wmma_bf16(const FragU& a, const FragU& b, v8f c) {
  return __builtin_amdgcn_wmma_f32_16x16x32_bf16(
      false, a.v, false, b.v, (short)0, c, false, false);
}

// ---------------- kernel 0: f32 -> bf16 (8 elems/thread) ----------------
__global__ __launch_bounds__(THREADS)
void k_convert(const float* __restrict__ x, unsigned short* __restrict__ xbf) {
  int i = blockIdx.x * THREADS + threadIdx.x;
  const float4* src = (const float4*)x;
  float4 a = src[i * 2 + 0];
  float4 b = src[i * 2 + 1];
  uint4 o;
  o.x = (unsigned)f32_to_bf16(a.x) | ((unsigned)f32_to_bf16(a.y) << 16);
  o.y = (unsigned)f32_to_bf16(a.z) | ((unsigned)f32_to_bf16(a.w) << 16);
  o.z = (unsigned)f32_to_bf16(b.x) | ((unsigned)f32_to_bf16(b.y) << 16);
  o.w = (unsigned)f32_to_bf16(b.z) | ((unsigned)f32_to_bf16(b.w) << 16);
  ((uint4*)xbf)[i] = o;
}

__global__ __launch_bounds__(THREADS)
void k_zero(float* __restrict__ w) {
  int i = blockIdx.x * THREADS + threadIdx.x;
  if (i < B_ * S_) w[i] = 0.f;
}

// Issue async copy of chunk c (kt = c>>2, dc = c&3): 128 rows x 256 dims bf16.
static __device__ __forceinline__ void issue_chunk(
    const unsigned short* __restrict__ xbf, unsigned ks_base,
    int b, int c, int tid) {
  int kt = c >> 2, dc = c & 3;
  const unsigned short* src = xbf + (size_t)(b * S_ + kt * KT) * D_ + dc * DC;
#pragma unroll
  for (int i = 0; i < 16; ++i) {
    int e = tid + i * THREADS;         // 16B chunk id; 32 chunks per row
    int r = e >> 5, cc = e & 31;
    async_cp16(ks_base + (unsigned)e * 16u,
               (unsigned long long)(uintptr_t)(src + (size_t)r * D_ + cc * 8));
  }
}

// 8 d-steps x 4 WMMAs on a staged 128x256 K chunk. Each A/B fragment feeds
// two WMMAs; 4 independent accumulators (no D->A/B hazards back-to-back).
// NOTE: Ks0 is a single LDS-rooted pointer + integer buffer offset so the
// address space infers to LDS (ds_load_b128), never FLAT.
static __device__ __forceinline__ void chunk_compute(
    const unsigned short* __restrict__ q0p,    // Q row base incl. d-chunk offset
    const unsigned short* __restrict__ Ks0,
    int buf_off,                               // (c&1)*KBUF_USH
    int kquad, int lr, int half, v8f (&acc)[2][2]) {
  const unsigned short* k0p = Ks0 + buf_off + (kquad * 32 + lr) * DC;
#pragma unroll
  for (int dd = 0; dd < DC; dd += 32) {
    FragU fa0, fa1, fb0, fb1;
    const uint4* pa0 = (const uint4*)(q0p + dd + 16 * half);
    const uint4* pa1 = (const uint4*)(q0p + 16 * D_ + dd + 16 * half);
    fa0.u4[0] = pa0[0]; fa0.u4[1] = pa0[1];
    fa1.u4[0] = pa1[0]; fa1.u4[1] = pa1[1];
    const uint4* pb0 = (const uint4*)(k0p + dd + 16 * half);
    const uint4* pb1 = (const uint4*)(k0p + 16 * DC + dd + 16 * half);
    fb0.u4[0] = pb0[0]; fb0.u4[1] = pb0[1];
    fb1.u4[0] = pb1[0]; fb1.u4[1] = pb1[1];

    acc[0][0] = wmma_bf16(fa0, fb0, acc[0][0]);
    acc[0][1] = wmma_bf16(fa0, fb1, acc[0][1]);
    acc[1][0] = wmma_bf16(fa1, fb0, acc[1][0]);
    acc[1][1] = wmma_bf16(fa1, fb1, acc[1][1]);
  }
}

// ------- kernel 1: fused 2-pass attention -> w[k] = sum_q attn[q,k] -------
__global__ __launch_bounds__(THREADS)
void k_attn(const unsigned short* __restrict__ xbf, float* __restrict__ wg) {
  extern __shared__ char smem[];
  unsigned short* Qs    = (unsigned short*)smem;                   // 64x1024 = 128K
  unsigned short* Ks0   = (unsigned short*)(smem + 131072);        // 2 x (128x256) = 128K
  float*          wpart = (float*)(smem + 262144);                 // 2048 f32
  float*          redM  = wpart + S_;                              // 8x32
  float*          redZ  = redM + 256;
  float*          mfin  = redZ + 256;                              // 64
  float*          zinv  = mfin + 64;

  const int tid  = threadIdx.x;
  const int wv   = tid >> 5;
  const int lane = tid & 31;
  const int lr   = lane & 15;
  const int half = lane >> 4;
  const int qpair = wv & 1;     // rows qpair*32 + sub*16
  const int kquad = wv >> 1;    // keys kquad*32 + t*16

  const int b  = blockIdx.x >> 5;
  const int q0 = (blockIdx.x & 31) * QT;

  const unsigned ksb0 = (unsigned)(uintptr_t)Ks0;
  const unsigned short* qrow = Qs + (qpair * 32 + lr) * D_;
  const v8f zv = {0.f, 0.f, 0.f, 0.f, 0.f, 0.f, 0.f, 0.f};

  {  // async stage Q tile (contiguous 128 KiB)
    const unsigned short* src = xbf + (size_t)(b * S_ + q0) * D_;
    unsigned qb = (unsigned)(uintptr_t)Qs;
#pragma unroll
    for (int i = 0; i < 32; ++i) {
      int e = tid + i * THREADS;
      async_cp16(qb + (unsigned)e * 16u,
                 (unsigned long long)(uintptr_t)(src + e * 8));
    }
  }
  for (int i = tid; i < S_; i += THREADS) wpart[i] = 0.f;

  float rml[2][8], rzl[2][8];
#pragma unroll
  for (int s = 0; s < 2; ++s)
#pragma unroll
    for (int v = 0; v < 8; ++v) { rml[s][v] = -__builtin_inff(); rzl[s][v] = 0.f; }

  v8f acc[2][2];

  // ---------------- pass 1: double-buffered chunk stream ----------------
  issue_chunk(xbf, ksb0, b, 0, tid);
  wait_async16();                       // drain Q copy; keep ASYNCcnt <= 32
  for (int c = 0; c < NCH; ++c) {
    const int dc = c & 3;
    if (c + 1 < NCH) {
      issue_chunk(xbf, ksb0 + (unsigned)(((c + 1) & 1) * (KBUF_USH * 2)),
                  b, c + 1, tid);       // prefetch next into other buffer
      wait_async16();                   // in-order: chunk c landed
    } else {
      wait_async0();
    }
    __syncthreads();
    if (dc == 0) { acc[0][0] = zv; acc[0][1] = zv; acc[1][0] = zv; acc[1][1] = zv; }
    chunk_compute(qrow + dc * DC, Ks0, (c & 1) * KBUF_USH, kquad, lr, half, acc);
    if (dc == 3) {                      // panel done: per-lane online update
#pragma unroll
      for (int s = 0; s < 2; ++s)
#pragma unroll
        for (int v = 0; v < 8; ++v) {
          float s0 = acc[s][0][v], s1 = acc[s][1][v];
          float mnew = fmaxf(rml[s][v], fmaxf(s0, s1));
          rzl[s][v] = rzl[s][v] * __expf(rml[s][v] - mnew)
                    + __expf(s0 - mnew) + __expf(s1 - mnew);
          rml[s][v] = mnew;
        }
    }
    __syncthreads();                    // release buffer for chunk c+2
  }

  // fold the 16 column-lanes once
#pragma unroll
  for (int s = 0; s < 2; ++s)
#pragma unroll
    for (int v = 0; v < 8; ++v) {
      float m = rml[s][v];
      m = fmaxf(m, __shfl_xor(m, 8, 32));
      m = fmaxf(m, __shfl_xor(m, 4, 32));
      m = fmaxf(m, __shfl_xor(m, 2, 32));
      m = fmaxf(m, __shfl_xor(m, 1, 32));
      float zz = rzl[s][v] * __expf(rml[s][v] - m);
      zz += __shfl_xor(zz, 8, 32);
      zz += __shfl_xor(zz, 4, 32);
      zz += __shfl_xor(zz, 2, 32);
      zz += __shfl_xor(zz, 1, 32);
      rml[s][v] = m;
      rzl[s][v] = zz;
    }
  if (lr == 0) {
#pragma unroll
    for (int s = 0; s < 2; ++s)
#pragma unroll
      for (int v = 0; v < 8; ++v) {
        int idx = wv * 32 + s * 16 + v + 8 * half;
        redM[idx] = rml[s][v];
        redZ[idx] = rzl[s][v];
      }
  }
  __syncthreads();
  if (tid < QT) {   // combine the 4 kquad-waves covering each query row
    int qp = tid >> 5, rl = tid & 31;
    float ms[4], zs[4], m = -__builtin_inff();
#pragma unroll
    for (int kq = 0; kq < 4; ++kq) {
      ms[kq] = redM[(qp + 2 * kq) * 32 + rl];
      zs[kq] = redZ[(qp + 2 * kq) * 32 + rl];
      m = fmaxf(m, ms[kq]);
    }
    float z = 0.f;
#pragma unroll
    for (int kq = 0; kq < 4; ++kq) z += zs[kq] * __expf(ms[kq] - m);
    mfin[tid] = m;
    zinv[tid] = 1.f / z;
  }
  __syncthreads();

  float fm[2][8], fz[2][8];
#pragma unroll
  for (int s = 0; s < 2; ++s)
#pragma unroll
    for (int v = 0; v < 8; ++v) {
      int row = qpair * 32 + s * 16 + v + 8 * half;
      fm[s][v] = mfin[row];
      fz[s][v] = zinv[row];
    }

  // ---------------- pass 2: w[k] += sum_q exp(s-m)/Z ----------------
  issue_chunk(xbf, ksb0, b, 0, tid);
  for (int c = 0; c < NCH; ++c) {
    const int dc = c & 3;
    if (c + 1 < NCH) {
      issue_chunk(xbf, ksb0 + (unsigned)(((c + 1) & 1) * (KBUF_USH * 2)),
                  b, c + 1, tid);
      wait_async16();
    } else {
      wait_async0();
    }
    __syncthreads();
    if (dc == 0) { acc[0][0] = zv; acc[0][1] = zv; acc[1][0] = zv; acc[1][1] = zv; }
    chunk_compute(qrow + dc * DC, Ks0, (c & 1) * KBUF_USH, kquad, lr, half, acc);
    if (dc == 3) {
      const int kt = c >> 2;
#pragma unroll
      for (int t = 0; t < 2; ++t) {
        float cs = 0.f;
#pragma unroll
        for (int s = 0; s < 2; ++s)
#pragma unroll
          for (int v = 0; v < 8; ++v)
            cs += __expf(acc[s][t][v] - fm[s][v]) * fz[s][v];
        cs += __shfl_xor(cs, 16, 32);          // fold the two 8-row halves
        if (half == 0)
          atomicAdd(&wpart[kt * KT + kquad * 32 + t * 16 + lr], cs);  // ds_add_f32
      }
    }
    __syncthreads();
  }
  for (int i = tid; i < S_; i += THREADS)
    atomicAdd(&wg[b * S_ + i], wpart[i]);      // global_atomic_add_f32
}

// -------- kernel 2: out[b,d] = (1/S) * sum_k w[b,k] * x[b,k,d] --------
__global__ __launch_bounds__(THREADS)
void k_pool(const float* __restrict__ w, const float* __restrict__ x,
            float* __restrict__ out) {
  int b = blockIdx.x >> 2;
  int d = (blockIdx.x & 3) * THREADS + threadIdx.x;
  const float* xb = x + (size_t)b * S_ * D_;
  const float* wb = w + b * S_;
  float acc = 0.f;
#pragma unroll 4
  for (int k = 0; k < S_; ++k)
    acc += wb[k] * xb[(size_t)k * D_ + d];
  out[b * D_ + d] = acc * (1.f / (float)S_);
}

extern "C" void kernel_launch(void* const* d_in, const int* in_sizes, int n_in,
                              void* d_out, int out_size, void* d_ws, size_t ws_size,
                              hipStream_t stream) {
  const float* x = (const float*)d_in[0];
  float* out = (float*)d_out;
  unsigned short* xbf = (unsigned short*)d_ws;
  float* wg = (float*)((char*)d_ws + XBF_BYTES);

  k_convert<<<(B_ * S_ * D_ / 8) / THREADS, THREADS, 0, stream>>>(x, xbf);
  k_zero<<<(B_ * S_ + THREADS - 1) / THREADS, THREADS, 0, stream>>>(wg);

  // 128K (Q) + 2x64K (K double buffer) + 8K (wpart) + 2K (red) + 512B (m/z)
  size_t smem = 131072 + 2 * 65536 + S_ * 4 + 256 * 4 * 2 + 64 * 4 * 2;
  k_attn<<<B_ * (S_ / QT), THREADS, smem, stream>>>(xbf, wg);

  k_pool<<<B_ * (D_ / THREADS), THREADS, 0, stream>>>(wg, x, out);
}
